// BitConv2dInfer_6330781795072
// MI455X (gfx1250) — 8-bit matrix operand order repaired and checked by reading, compiled, not run on hardware
//
#include <hip/hip_runtime.h>

typedef int v8i __attribute__((ext_vector_type(8)));
typedef int v4i __attribute__((ext_vector_type(4)));
typedef int v2i __attribute__((ext_vector_type(2)));

#define NIMG  32
#define CIN   256
#define COUT  256
#define H     56
#define W     56
#define HP    58
#define WP    58
#define KTOT  2304
#define NSTEPS 36
#define TOTAL_SPATIAL (NIMG*H*W)

#define XQ_BYTES (NIMG*HP*WP*CIN)
#define WT_OFF   XQ_BYTES
#define WT_BYTES (COUT*KTOT)

__device__ __forceinline__ void async_g2l_b128(unsigned lds_addr, const void* gptr) {
  asm volatile("global_load_async_to_lds_b128 %0, %1, off"
               :: "v"(lds_addr), "v"(gptr)
               : "memory");
}
__device__ __forceinline__ void wait_async0() {
  asm volatile("s_wait_asynccnt 0x0" ::: "memory");
}

__global__ void zero_ws_kernel(int4* __restrict__ p, int n4) {
  int i = blockIdx.x * blockDim.x + threadIdx.x;
  if (i < n4) {
    int4 z; z.x = 0; z.y = 0; z.z = 0; z.w = 0;
    p[i] = z;
  }
}

__global__ void quant_kernel(const float* __restrict__ x,
                             signed char* __restrict__ xq,
                             const float* __restrict__ act_s_p,
                             int total) {
  int idx = blockIdx.x * blockDim.x + threadIdx.x;
  if (idx >= total) return;
  const float as = *act_s_p;
  float v = x[idx];
  v = fminf(fmaxf(v, -1.0f), 1.0f);
  float q = rintf(v / as);
  q = fminf(fmaxf(q, -127.0f), 127.0f);
  const int iw = idx % W;
  int t = idx / W;
  const int ih = t % H;
  t = t / H;
  const int ci = t % CIN;
  const int n  = t / CIN;
  xq[(((size_t)(n * HP + ih + 1) * WP) + (iw + 1)) * CIN + ci] = (signed char)(int)q;
}

__global__ void wtrans_kernel(const signed char* __restrict__ w,
                              signed char* __restrict__ wt) {
  int i = blockIdx.x * blockDim.x + threadIdx.x;
  if (i >= COUT * KTOT) return;
  const int ci = i % 256;
  const int p  = (i / 256) % 9;
  const int co = i / KTOT;
  wt[i] = w[(size_t)co * KTOT + ci * 9 + p];
}

__global__ __launch_bounds__(256)
void bitconv_gemm_kernel(const signed char* __restrict__ Wt,
                         const signed char* __restrict__ Xq,
                         const float* __restrict__ s,
                         const float* __restrict__ bias,
                         const float* __restrict__ act_s_p,
                         float* __restrict__ out) {
  __shared__ __attribute__((aligned(16))) char lds[49152];

  const int tid   = threadIdx.x;
  const int lane  = tid & 31;
  const int wv    = tid >> 5;
  const int waveM = wv >> 2;
  const int waveN = wv & 3;
  const int l16   = lane & 15;
  const int hlf   = lane >> 4;

  const int bN = blockIdx.x;
  const int bM = blockIdx.y;

  const int aRow  = tid >> 1;
  const int aHalf = (tid & 1) * 32;
  const signed char* aPtr = Wt + (size_t)(bM * 128 + aRow) * KTOT + aHalf;

  const int j  = bN * 256 + tid;
  const int ow = j % W;
  const int jt = j / W;
  const int oh = jt % H;
  const int ni = jt / H;
  const signed char* bPtr =
      Xq + ((size_t)((ni * HP + oh) * WP + ow)) * CIN;

  const unsigned ldsBase = (unsigned)(size_t)&lds[0];
  const unsigned stA = ldsBase + aRow * 64 + aHalf;
  const unsigned stB = ldsBase + 16384 + tid * 64;

  async_g2l_b128(stA,      aPtr);
  async_g2l_b128(stA + 16, aPtr + 16);
  async_g2l_b128(stB,      bPtr);
  async_g2l_b128(stB + 16, bPtr + 16);
  async_g2l_b128(stB + 32, bPtr + 32);
  async_g2l_b128(stB + 48, bPtr + 48);

  v8i acc[4][4];
#pragma unroll
  for (int a = 0; a < 4; ++a)
#pragma unroll
    for (int b = 0; b < 4; ++b)
      acc[a][b] = (v8i){0, 0, 0, 0, 0, 0, 0, 0};

  for (int ks = 0; ks < NSTEPS; ++ks) {
    const int cur = ks & 1;

    wait_async0();
    __syncthreads();

    if (ks < NSTEPS - 1) {
      const int sn  = ks + 1;
      const int nb  = sn & 1;
      const int pp  = sn >> 2;
      const int ci0 = (sn & 3) << 6;
      const int kh  = pp / 3;
      const int kw  = pp - kh * 3;
      const signed char* ag = aPtr + sn * 64;
      const unsigned   sa = stA + nb * 8192;
      async_g2l_b128(sa,      ag);
      async_g2l_b128(sa + 16, ag + 16);
      const int ob = (kh * WP + kw) * CIN + ci0;
      const signed char* bg = bPtr + ob;
      const unsigned   sb = stB + nb * 16384;
      async_g2l_b128(sb,      bg);
      async_g2l_b128(sb + 16, bg + 16);
      async_g2l_b128(sb + 32, bg + 32);
      async_g2l_b128(sb + 48, bg + 48);
    }

    const char* Abase = lds + cur * 8192;
    const char* Bbase = lds + 16384 + cur * 16384;

    v8i af[4];
#pragma unroll
    for (int a = 0; a < 4; ++a) {
      const char* ap = Abase + (waveM * 64 + a * 16 + l16) * 64 + hlf * 16;
      v4i q0 = *(const v4i*)(ap);
      v4i q1 = *(const v4i*)(ap + 32);
      af[a] = (v8i){q0.x, q0.y, q0.z, q0.w, q1.x, q1.y, q1.z, q1.w};
    }
    v8i bf[4];
#pragma unroll
    for (int b = 0; b < 4; ++b) {
      const char* bp = Bbase + (waveN * 64 + b * 16 + l16) * 64 + hlf * 16;
      v4i r0 = *(const v4i*)(bp);
      v4i r1 = *(const v4i*)(bp + 32);
      bf[b] = (v8i){r0.x, r0.y, r0.z, r0.w, r1.x, r1.y, r1.z, r1.w};
    }

#pragma unroll
    for (int a = 0; a < 4; ++a)
#pragma unroll
      for (int b = 0; b < 4; ++b)
        acc[a][b] = __builtin_amdgcn_wmma_i32_16x16x64_iu8(
            true, af[a], true, bf[b], acc[a][b], false, false);
  }

  const float as  = *act_s_p;
  const int coB = bM * 128 + waveM * 64;
  const int jB  = bN * 256 + waveN * 64;

  int baseb[4];
#pragma unroll
  for (int b = 0; b < 4; ++b) {
    const int jj  = jB + b * 16 + l16;
    const int oww = jj % W;
    const int jjt = jj / W;
    const int ohh = jjt % H;
    const int nn  = jjt / H;
    baseb[b] = nn * (COUT * H * W) + ohh * W + oww;
  }

#pragma unroll
  for (int a = 0; a < 4; ++a) {
#pragma unroll
    for (int v = 0; v < 8; ++v) {
      const int co = coB + a * 16 + hlf * 8 + v;
      const float sc = as * s[co];
      const float bi = bias[co];
#pragma unroll
      for (int b = 0; b < 4; ++b) {
        out[(size_t)baseb[b] + (size_t)co * (H * W)] =
            fmaf((float)acc[a][b][v], sc, bi);
      }
    }
  }
}

extern "C" void kernel_launch(void* const* d_in, const int* in_sizes, int n_in,
                              void* d_out, int out_size, void* d_ws, size_t ws_size,
                              hipStream_t stream) {
  const float*       x    = (const float*)d_in[0];
  const signed char* wq   = (const signed char*)d_in[1];
  const float*       s    = (const float*)d_in[2];
  const float*       bias = (const float*)d_in[3];
  const float*       as   = (const float*)d_in[4];
  float* out = (float*)d_out;

  signed char* xq = (signed char*)d_ws;
  signed char* wt = (signed char*)d_ws + WT_OFF;

  {
    const int n4 = XQ_BYTES / 16;
    zero_ws_kernel<<<(n4 + 255) / 256, 256, 0, stream>>>((int4*)d_ws, n4);
  }
  {
    const int total = NIMG * CIN * H * W;
    quant_kernel<<<(total + 255) / 256, 256, 0, stream>>>(x, xq, as, total);
  }
  {
    const int total = COUT * KTOT;
    wtrans_kernel<<<(total + 255) / 256, 256, 0, stream>>>(wq, wt);
  }
  {
    dim3 grid(TOTAL_SPATIAL / 256, COUT / 128);
    bitconv_gemm_kernel<<<grid, 256, 0, stream>>>(wt, xq, s, bias, as, out);
  }
  (void)in_sizes; (void)n_in; (void)out_size; (void)ws_size;
}
